// SA_Layer_47983374631061
// MI455X (gfx1250) — matrix operand order repaired and checked by reading, not compiled, not run on hardware
//
#include <hip/hip_runtime.h>
#include <math.h>

#define B_  128
#define C_  256
#define N_  4096
#define CQ_ 64
#define TN_ 128           // n-points per workgroup in energy kernel
#define LDSROW_ 132       // 128 f32 + 4 f32 TDM pad -> conflict-free transpose reads

typedef __attribute__((ext_vector_type(16))) _Float16 v16h;
typedef __attribute__((ext_vector_type(8)))  float    v8f;
typedef __attribute__((ext_vector_type(4)))  float    f4v;
typedef __attribute__((ext_vector_type(4)))  unsigned int v4u;
typedef __attribute__((ext_vector_type(4)))  int      v4i;
typedef __attribute__((ext_vector_type(8)))  int      v8i;

// ---------------------------------------------------------------------------
// TDM: DMA a 32-row x 128-col f32 tile (row stride N_ elements) from global
// into LDS at lds_byte_off, inserting 16B of pad after every 512B row so the
// LDS row stride is 132 floats. One instruction, tracked by TENSORcnt.
// ---------------------------------------------------------------------------
__device__ __forceinline__ void tdm_load_tile(const float* gsrc, unsigned lds_byte_off) {
    unsigned long long ga = (unsigned long long)(uintptr_t)gsrc;
    v4u g0;
    g0.x = 1u;                                              // count=1, user mode
    g0.y = lds_byte_off;                                    // LDS byte address
    g0.z = (unsigned)(ga & 0xFFFFFFFFu);                    // global_addr[31:0]
    g0.w = (unsigned)((ga >> 32) & 0x01FFFFFFu) | (2u << 30); // addr[56:32], type=2

    v8i g1;
    // data_size=2 (4B) | pad_enable | pad_interval=6 (512B) | pad_amount=3 (4 dwords)
    g1[0] = (int)((2u << 16) | (1u << 20) | (6u << 22) | (3u << 25));
    const unsigned dim0 = 0x40000000u, dim1 = 0x40000000u;  // huge: never OOB-clamped
    g1[1] = (int)((dim0 & 0xFFFFu) << 16);                  // tensor_dim0[15:0]
    g1[2] = (int)((dim0 >> 16) | ((dim1 & 0xFFFFu) << 16)); // dim0[31:16], dim1[15:0]
    g1[3] = (int)((dim1 >> 16) | ((unsigned)TN_ << 16));    // dim1[31:16], tile_dim0=128
    g1[4] = (int)32;                                        // tile_dim1=32, tile_dim2=0
    g1[5] = (int)N_;                                        // tensor_dim0_stride = 4096
    g1[6] = 0;
    g1[7] = 0;

    v4i z4 = {0, 0, 0, 0};
#if defined(__clang_major__) && (__clang_major__ >= 23)
    v8i z8 = {0, 0, 0, 0, 0, 0, 0, 0};
    __builtin_amdgcn_tensor_load_to_lds(g0, g1, z4, z4, z8, 0);
#else
    __builtin_amdgcn_tensor_load_to_lds(g0, g1, z4, z4, 0);
#endif
}

// ---------------------------------------------------------------------------
// Kernel 0: pack wq (CQ x C, f32) into per-WMMA-B-fragment f16 layout.
// Fragment (ot in 0..3, ks in 0..7): B is 32(K) x 16(o).
// Lane L holds column o = ot*16 + (L%16); half slot e holds k = ks*32 + 16*(e/8) + 8*(L/16) + e%8.
// ---------------------------------------------------------------------------
__global__ __launch_bounds__(256) void pack_wq_kernel(const float* __restrict__ wq,
                                                      _Float16* __restrict__ wqpack) {
    int idx = blockIdx.x * 256 + threadIdx.x;   // 0 .. 16383
    int e   = idx & 15;
    int L   = (idx >> 4) & 31;
    int f   = idx >> 9;                         // 0..31
    int ot  = f >> 3;
    int ks  = f & 7;
    wqpack[idx] = (_Float16)wq[(ot * 16 + (L & 15)) * C_ + ks * 32 + ((e >> 3) << 4) + ((L >> 4) << 3) + (e & 7)];
}

// ---------------------------------------------------------------------------
// Kernel 1: tk[b,o] = tanh(wk[o,:] . x_key[b,:] + bk[o])
// ---------------------------------------------------------------------------
__global__ __launch_bounds__(64) void key_proj_kernel(const float* __restrict__ x_key,
                                                      const float* __restrict__ wk,
                                                      const float* __restrict__ bk,
                                                      float* __restrict__ tk) {
    int b = blockIdx.x;
    int o = threadIdx.x;
    const float* xk = x_key + b * C_;
    const float* w  = wk + o * C_;
    float s = bk[o];
    for (int c = 0; c < C_; ++c) s += w[c] * xk[c];
    tk[b * CQ_ + o] = tanhf(s);
}

// ---------------------------------------------------------------------------
// Kernel 2: energy[b,n] = sum_o tanh( (wq x)[b,n,o] + bq[o] ) * tk[b,o]
// One workgroup (8 waves) per (batch, 128-pt tile). TDM double-buffers the
// x K-slab into LDS; each wave owns one 16-pt subtile and runs
// v_wmma_f32_16x16x32_f16 over 4 o-tiles x 8 K-steps.
// ---------------------------------------------------------------------------
__global__ __launch_bounds__(256) void energy_kernel(const float* __restrict__ x,
                                                     const _Float16* __restrict__ wqpack,
                                                     const float* __restrict__ bq,
                                                     const float* __restrict__ tk,
                                                     float* __restrict__ energy) {
    __shared__ float xs[2][32 * LDSROW_];   // 2 x 16.5 KB double buffer (TDM-padded rows)

    const int ntiles = N_ / TN_;            // 32
    int b    = blockIdx.x / ntiles;
    int n0   = (blockIdx.x % ntiles) * TN_;
    int tid  = threadIdx.x;
    int wave = tid >> 5;
    int lane = tid & 31;

    const float* xb = x + (size_t)b * C_ * N_ + n0;

    v8f acc[4];
    #pragma unroll
    for (int ot = 0; ot < 4; ++ot)
        #pragma unroll
        for (int r = 0; r < 8; ++r) acc[ot][r] = 0.0f;

    // prologue: DMA K-slab 0 into buffer 0
    if (wave == 0) tdm_load_tile(xb, (unsigned)(uintptr_t)&xs[0][0]);

    int m    = lane & 15;        // A row within subtile
    int kg   = lane >> 4;        // lane group selects K range
    int ncol = wave * 16 + m;    // column in staged tile

    for (int ks = 0; ks < 8; ++ks) {
        __syncthreads();   // all waves done reading the buffer the next DMA overwrites
        if (wave == 0) {
            if (ks < 7) {
                tdm_load_tile(xb + (size_t)(ks + 1) * 32 * N_,
                              (unsigned)(uintptr_t)&xs[(ks + 1) & 1][0]);
                __builtin_amdgcn_s_wait_tensorcnt((short)1);   // slab ks landed
            } else {
                __builtin_amdgcn_s_wait_tensorcnt((short)0);
            }
        }
        __syncthreads();   // slab ks visible to all waves

        const float* xt = &xs[ks & 1][0];

        // Build 16x32 f16 A-fragment (ISA 7.12.2 layout): transpose via LDS reads.
        v16h a;
        #pragma unroll
        for (int v = 0; v < 4; ++v) {
            int k0 = kg * 8 + 2 * v;          // VGPR v   : K = k0, k0+1
            int k1 = 16 + kg * 8 + 2 * v;     // VGPR v+4 : K = k1, k1+1
            a[2 * v]         = (_Float16)xt[(k0)     * LDSROW_ + ncol];
            a[2 * v + 1]     = (_Float16)xt[(k0 + 1) * LDSROW_ + ncol];
            a[8 + 2 * v]     = (_Float16)xt[(k1)     * LDSROW_ + ncol];
            a[8 + 2 * v + 1] = (_Float16)xt[(k1 + 1) * LDSROW_ + ncol];
        }

        #pragma unroll
        for (int ot = 0; ot < 4; ++ot) {
            v16h bfrag = ((const v16h*)wqpack)[(ot * 8 + ks) * 32 + lane];
            acc[ot] = __builtin_amdgcn_wmma_f32_16x16x32_f16(
                false, a, false, bfrag, (short)0, acc[ot], false, false);
        }
    }

    // Epilogue: tanh-gate, weight by tk, reduce over o (16 lanes x 4 o-tiles).
    int ocol = lane & 15;
    int lg   = lane >> 4;
    float bqv[4], tkv[4];
    #pragma unroll
    for (int ot = 0; ot < 4; ++ot) {
        bqv[ot] = bq[ot * 16 + ocol];
        tkv[ot] = tk[b * CQ_ + ot * 16 + ocol];
    }
    int nbase = n0 + wave * 16 + lg * 8;     // D row M = r + 8*lg
    #pragma unroll
    for (int r = 0; r < 8; ++r) {
        float e = 0.0f;
        #pragma unroll
        for (int ot = 0; ot < 4; ++ot)
            e += tanhf(acc[ot][r] + bqv[ot]) * tkv[ot];
        e += __shfl_xor(e, 1, 32);
        e += __shfl_xor(e, 2, 32);
        e += __shfl_xor(e, 4, 32);
        e += __shfl_xor(e, 8, 32);
        if (ocol == 0) energy[b * N_ + nbase + r] = e;
    }
}

// ---------------------------------------------------------------------------
// Kernel 3: softmax over N per batch -> attention (written into d_out slice)
// ---------------------------------------------------------------------------
__global__ __launch_bounds__(256) void softmax_kernel(const float* __restrict__ energy,
                                                      float* __restrict__ att) {
    __shared__ float red[256];
    int b = blockIdx.x, t = threadIdx.x;
    const float* e = energy + (size_t)b * N_;

    float mx = -3.4e38f;
    for (int i = t; i < N_; i += 256) mx = fmaxf(mx, e[i]);
    red[t] = mx; __syncthreads();
    for (int s = 128; s > 0; s >>= 1) { if (t < s) red[t] = fmaxf(red[t], red[t + s]); __syncthreads(); }
    mx = red[0]; __syncthreads();

    float sum = 0.0f;
    for (int i = t; i < N_; i += 256) sum += __expf(e[i] - mx);
    red[t] = sum; __syncthreads();
    for (int s = 128; s > 0; s >>= 1) { if (t < s) red[t] += red[t + s]; __syncthreads(); }
    float inv = 1.0f / red[0];

    float* a = att + (size_t)b * N_;
    for (int i = t; i < N_; i += 256) a[i] = __expf(e[i] - mx) * inv;
}

// ---------------------------------------------------------------------------
// Kernel 4: x_ctx[b,c] = sum_n x[b,c,n] * att[b,n]   (the 512 MB HBM pass)
// One wave per (b,c); non-temporal float4 streaming loads for x (never fits
// L2), regular-temporal for the 2 MB attention vector (stays L2-hot).
// ---------------------------------------------------------------------------
__global__ __launch_bounds__(256) void context_kernel(const float* __restrict__ x,
                                                      const float* __restrict__ att,
                                                      float* __restrict__ xctx) {
    int gw   = blockIdx.x * 8 + (threadIdx.x >> 5);  // global wave id = (b,c) pair
    int lane = threadIdx.x & 31;
    int b = gw / C_, c = gw % C_;
    const f4v* xr = (const f4v*)(x + ((size_t)b * C_ + c) * N_);
    const f4v* ar = (const f4v*)(att + (size_t)b * N_);
    float s = 0.0f;
    for (int i = lane; i < N_ / 4; i += 32) {
        f4v xv = __builtin_nontemporal_load(xr + i);
        f4v av = ar[i];
        s += xv[0] * av[0] + xv[1] * av[1] + xv[2] * av[2] + xv[3] * av[3];
    }
    #pragma unroll
    for (int mmask = 16; mmask > 0; mmask >>= 1) s += __shfl_xor(s, mmask, 32);
    if (lane == 0) xctx[b * C_ + c] = s;
}

// ---------------------------------------------------------------------------
// Kernel 5: per batch: x_r = wv.xctx + bv ; diff = x_key - x_r ;
//           t = wt.diff + bt   (tiny chained matvecs, fused via LDS)
// ---------------------------------------------------------------------------
__global__ __launch_bounds__(256) void trans_kernel(const float* __restrict__ x_key,
                                                    const float* __restrict__ xctx,
                                                    const float* __restrict__ wv,
                                                    const float* __restrict__ bv,
                                                    const float* __restrict__ wt,
                                                    const float* __restrict__ bt,
                                                    float* __restrict__ tbuf) {
    __shared__ float ctx[C_];
    __shared__ float diff[C_];
    int b = blockIdx.x, t = threadIdx.x;
    ctx[t] = xctx[b * C_ + t];
    __syncthreads();
    float s = bv[t];
    const float* w = wv + t * C_;
    for (int c = 0; c < C_; ++c) s += w[c] * ctx[c];
    diff[t] = x_key[b * C_ + t] - s;
    __syncthreads();
    float s2 = bt[t];
    const float* w2 = wt + t * C_;
    for (int c = 0; c < C_; ++c) s2 += w2[c] * diff[c];
    tbuf[b * C_ + t] = s2;
}

// ---------------------------------------------------------------------------
// Kernel 6: training-mode BatchNorm over batch dim + ReLU + residual.
// ---------------------------------------------------------------------------
__global__ __launch_bounds__(128) void bn_kernel(const float* __restrict__ tbuf,
                                                 const float* __restrict__ x_key,
                                                 const float* __restrict__ gamma,
                                                 const float* __restrict__ beta,
                                                 float* __restrict__ out) {
    __shared__ float rs[128], rs2[128];
    int c = blockIdx.x, b = threadIdx.x;
    float v = tbuf[b * C_ + c];
    rs[b] = v; rs2[b] = v * v;
    __syncthreads();
    for (int s = 64; s > 0; s >>= 1) {
        if (b < s) { rs[b] += rs[b + s]; rs2[b] += rs2[b + s]; }
        __syncthreads();
    }
    float mean = rs[0] * (1.0f / B_);
    float var  = rs2[0] * (1.0f / B_) - mean * mean;   // biased, torch-BN training
    float g  = gamma[c] * rsqrtf(var + 1e-5f);
    float bn = (v - mean) * g + beta[c];
    out[b * C_ + c] = x_key[b * C_ + c] + fmaxf(bn, 0.0f);
}

// ---------------------------------------------------------------------------
extern "C" void kernel_launch(void* const* d_in, const int* in_sizes, int n_in,
                              void* d_out, int out_size, void* d_ws, size_t ws_size,
                              hipStream_t stream) {
    (void)in_sizes; (void)n_in; (void)out_size; (void)ws_size;

    const float* x     = (const float*)d_in[0];   // (B, C, N)
    const float* x_key = (const float*)d_in[1];   // (B, C, 1)
    const float* wq    = (const float*)d_in[2];   // (CQ, C)
    const float* bq    = (const float*)d_in[3];   // (CQ)
    const float* wk    = (const float*)d_in[4];   // (CQ, C)
    const float* bk    = (const float*)d_in[5];   // (CQ)
    const float* wv    = (const float*)d_in[6];   // (C, C)
    const float* bv    = (const float*)d_in[7];   // (C)
    const float* wt    = (const float*)d_in[8];   // (C, C)
    const float* bt    = (const float*)d_in[9];   // (C)
    const float* gamma = (const float*)d_in[10];  // (C)
    const float* beta  = (const float*)d_in[11];  // (C)

    float* out_final = (float*)d_out;             // (B, C) = 32768 floats
    float* out_att   = (float*)d_out + B_ * C_;   // (B, N) = 524288 floats

    // Workspace layout (byte offsets)
    char* ws = (char*)d_ws;
    _Float16* wqpack = (_Float16*)(ws);                         //  32 KB
    float*    tk     = (float*)(ws + 32768);                    //  32 KB
    float*    energy = (float*)(ws + 65536);                    //   2 MB
    float*    xctx   = (float*)(ws + 65536 + 2097152);          // 128 KB
    float*    tbuf   = (float*)(ws + 65536 + 2097152 + 131072); // 128 KB

    pack_wq_kernel<<<64, 256, 0, stream>>>(wq, wqpack);
    key_proj_kernel<<<B_, 64, 0, stream>>>(x_key, wk, bk, tk);
    energy_kernel<<<B_ * (N_ / TN_), 256, 0, stream>>>(x, wqpack, bq, tk, energy);
    softmax_kernel<<<B_, 256, 0, stream>>>(energy, out_att);
    context_kernel<<<(B_ * C_) / 8, 256, 0, stream>>>(x, out_att, xctx);
    trans_kernel<<<B_, 256, 0, stream>>>(x_key, xctx, wv, bv, wt, bt, tbuf);
    bn_kernel<<<C_, 128, 0, stream>>>(tbuf, x_key, gamma, beta, out_final);
}
